// SSD_CON_11398843203941
// MI455X (gfx1250) — matrix operand order repaired and checked by reading, not compiled, not run on hardware
//
#include <hip/hip_runtime.h>
#include <hip/hip_bf16.h>

// ---------------------------------------------------------------------------
// SSD forward (main + flip path) for MI455X / gfx1250.
// Implicit-GEMM convolutions on v_wmma_f32_16x16x32_f16 (wave32),
// 2x2 register-blocked: one wave computes a 32x32 (Cout x pos) tile with
// 4 accumulators -> 2 global_load_b128 per WMMA from L2-resident data.
// Activations: NHWC f16.  Weights: packed [Cout][ky][kx][Cin] f16.
// ---------------------------------------------------------------------------

typedef __attribute__((ext_vector_type(16))) _Float16 v16h;
typedef __attribute__((ext_vector_type(8)))  _Float16 v8h;
typedef __attribute__((ext_vector_type(8)))  float    v8f;

__device__ __forceinline__ v16h cat8(v8h lo, v8h hi) {
  return __builtin_shufflevector(lo, hi, 0,1,2,3,4,5,6,7,
                                         8,9,10,11,12,13,14,15);
}
__device__ __forceinline__ v8f wmma16(v16h a, v16h b, v8f c) {
  return __builtin_amdgcn_wmma_f32_16x16x32_f16(false, a, false, b,
                                                (short)0, c, false, false);
}

// ---------------------------------------------------------------------------
// One wave accumulates a 32x32 (Cout x positions) tile (2x2 WMMA subtiles).
//   M = Cout, N = Hout*Wout (per batch), K = KHW*KHW*Cin  (k = (ky,kx,ci))
// CDNA5 f16 fragment layouts (CDNA5 ISA):
//   A 16x32: lanes 0-15: M=lane,   K = {0..7}U{16..23}   (two 8-half runs)
//            lanes16-31: M=lane-16,K = {8..15}U{24..31}
//   B 32x16: lanes 0-15: N=lane,   K = {0..7}U{16..23}   (two 8-half runs)
//            lanes16-31: N=lane-16,K = {8..15}U{24..31}
//   D 16x16: VGPR r: lanes 0-15 -> M=r, N=lane; lanes16-31 -> M=r+8
// Cin % 32 == 0 for every conv here, so all runs are contiguous memory.
// OOB rows/cols are clamped; their results are computed but never stored.
// ---------------------------------------------------------------------------
template <int KHW>
__device__ __forceinline__ void conv_tile_mm_2x2(
    const _Float16* __restrict__ inb,    // this batch, NHWC: [H][W][Cin]
    const _Float16* __restrict__ wp,     // packed f16:  [Cout][KHW][KHW][Cin]
    const _Float16* __restrict__ guard,  // >= Cin+32 zero halves
    int Cin, int Cout, int H, int W, int Wout,
    int stride, int pad, int flip_in,
    int m0, int n0, int npos, int lane,
    v8f& acc00, v8f& acc01, v8f& acc10, v8f& acc11)
{
  const int lan15 = lane & 15;
  const int K     = KHW * KHW * Cin;

  int ar0 = m0 + lan15;       if (ar0 >= Cout) ar0 = Cout - 1;
  int ar1 = m0 + 16 + lan15;  if (ar1 >= Cout) ar1 = Cout - 1;
  const _Float16* __restrict__ arow0 = wp + (size_t)ar0 * K;
  const _Float16* __restrict__ arow1 = wp + (size_t)ar1 * K;
  const int a_off = (lane < 16) ? 0 : 8;
  const int b_off = (lane < 16) ? 0 : 8;

  int bc0 = n0 + lan15;       if (bc0 >= npos) bc0 = 0;
  int bc1 = n0 + 16 + lan15;  if (bc1 >= npos) bc1 = 0;
  const int oy0 = bc0 / Wout, ox0 = bc0 - oy0 * Wout;
  const int oy1 = bc1 / Wout, ox1 = bc1 - oy1 * Wout;

#pragma unroll
  for (int cell = 0; cell < KHW * KHW; ++cell) {
    const int ky = cell / KHW;
    const int kx = cell - ky * KHW;

    const int iy0 = oy0 * stride + ky - pad;
    const int ix0 = ox0 * stride + kx - pad;
    const bool v0 = (iy0 >= 0) & (iy0 < H) & (ix0 >= 0) & (ix0 < W);
    const int sx0 = flip_in ? (W - 1 - ix0) : ix0;
    const _Float16* __restrict__ bb0 =
        v0 ? (inb + ((size_t)iy0 * W + sx0) * Cin) : guard;

    const int iy1 = oy1 * stride + ky - pad;
    const int ix1 = ox1 * stride + kx - pad;
    const bool v1 = (iy1 >= 0) & (iy1 < H) & (ix1 >= 0) & (ix1 < W);
    const int sx1 = flip_in ? (W - 1 - ix1) : ix1;
    const _Float16* __restrict__ bb1 =
        v1 ? (inb + ((size_t)iy1 * W + sx1) * Cin) : guard;

    const _Float16* __restrict__ ab0 = arow0 + (size_t)cell * Cin;
    const _Float16* __restrict__ ab1 = arow1 + (size_t)cell * Cin;
    __builtin_prefetch(ab0 + Cin, 0, 1);   // global_prefetch_b8
    __builtin_prefetch(ab1 + Cin, 0, 1);

    for (int c0 = 0; c0 < Cin; c0 += 32) {
      v16h A0 = cat8(*(const v8h*)(ab0 + c0 + a_off),
                     *(const v8h*)(ab0 + c0 + a_off + 16));
      v16h A1 = cat8(*(const v8h*)(ab1 + c0 + a_off),
                     *(const v8h*)(ab1 + c0 + a_off + 16));
      v16h B0 = cat8(*(const v8h*)(bb0 + c0 + b_off),
                     *(const v8h*)(bb0 + c0 + b_off + 16));
      v16h B1 = cat8(*(const v8h*)(bb1 + c0 + b_off),
                     *(const v8h*)(bb1 + c0 + b_off + 16));
      acc00 = wmma16(A0, B0, acc00);
      acc01 = wmma16(A0, B1, acc01);
      acc10 = wmma16(A1, B0, acc10);
      acc11 = wmma16(A1, B1, acc11);
    }
  }
}

// ---------------------------------------------------------------------------
// Extras conv: NHWC f16 in -> ReLU -> NHWC f16 out. 4 waves per block,
// one 32x32 tile per wave; grid.y = batch.  Extras have Cout % 16 == 0,
// so each subtile store is a single packed b128 per lane.
// ---------------------------------------------------------------------------
template <int KHW>
__global__ __launch_bounds__(128) void conv_relu_f16(
    const _Float16* __restrict__ in, const _Float16* __restrict__ wp,
    const float* __restrict__ bias, const _Float16* __restrict__ guard,
    _Float16* __restrict__ out,
    int Cin, int Cout, int H, int W, int Hout, int Wout,
    int stride, int pad, int flip_in)
{
  const int lane  = threadIdx.x & 31;
  const int lan15 = lane & 15;
  const int wave  = threadIdx.x >> 5;
  const int npos   = Hout * Wout;
  const int tilesN = (npos + 31) >> 5;
  const int tilesM = (Cout + 31) >> 5;
  const int tile = blockIdx.x * 4 + wave;
  if (tile >= tilesM * tilesN) return;   // wave-uniform
  const int tm = tile / tilesN;
  const int tn = tile - tm * tilesN;
  const int b = blockIdx.y;

  const _Float16* inb = in + (size_t)b * H * W * Cin;
  v8f a00 = {}, a01 = {}, a10 = {}, a11 = {};
  conv_tile_mm_2x2<KHW>(inb, wp, guard, Cin, Cout, H, W, Wout,
                        stride, pad, flip_in, tm * 32, tn * 32, npos, lane,
                        a00, a01, a10, a11);

  const int mrow = (lane < 16) ? 0 : 8;
#pragma unroll
  for (int mi = 0; mi < 2; ++mi) {
#pragma unroll
    for (int nj = 0; nj < 2; ++nj) {
      v8f acc = (mi == 0) ? (nj == 0 ? a00 : a01) : (nj == 0 ? a10 : a11);
      int d_col = tn * 32 + nj * 16 + lan15;
      int mbase = tm * 32 + mi * 16 + mrow;
      if (d_col < npos && mbase < Cout) {
        _Float16* op = out + ((size_t)b * npos + d_col) * Cout + mbase;
        v8h pk;
#pragma unroll
        for (int r = 0; r < 8; ++r)
          pk[r] = (_Float16)fmaxf(acc[r] + bias[mbase + r], 0.0f);
        *(v8h*)op = pk;   // 16B packed store (Cout%16==0 -> in range, aligned)
      }
    }
  }
}

// ---------------------------------------------------------------------------
// Head conv (3x3, s1, p1): NHWC f16 in -> fp32 out, written into the
// concatenated BHWC flatten (optionally W-flipped on read and/or write).
// outbase[b*batch_stride + src_off + (y*F + x_out)*Cout + m]
// ---------------------------------------------------------------------------
__global__ __launch_bounds__(128) void head_conv_f32(
    const _Float16* __restrict__ in, const _Float16* __restrict__ wp,
    const float* __restrict__ bias, const _Float16* __restrict__ guard,
    float* __restrict__ outbase,
    int Cin, int Cout, int F, int batch_stride, int src_off,
    int flip_in, int flip_back)
{
  const int lane  = threadIdx.x & 31;
  const int lan15 = lane & 15;
  const int wave  = threadIdx.x >> 5;
  const int npos   = F * F;
  const int tilesN = (npos + 31) >> 5;
  const int tilesM = (Cout + 31) >> 5;
  const int tile = blockIdx.x * 4 + wave;
  if (tile >= tilesM * tilesN) return;
  const int tm = tile / tilesN;
  const int tn = tile - tm * tilesN;
  const int b = blockIdx.y;

  const _Float16* inb = in + (size_t)b * npos * Cin;
  v8f a00 = {}, a01 = {}, a10 = {}, a11 = {};
  conv_tile_mm_2x2<3>(inb, wp, guard, Cin, Cout, F, F, F,
                      1, 1, flip_in, tm * 32, tn * 32, npos, lane,
                      a00, a01, a10, a11);

  const int mrow = (lane < 16) ? 0 : 8;
#pragma unroll
  for (int mi = 0; mi < 2; ++mi) {
#pragma unroll
    for (int nj = 0; nj < 2; ++nj) {
      v8f acc = (mi == 0) ? (nj == 0 ? a00 : a01) : (nj == 0 ? a10 : a11);
      int d_col = tn * 32 + nj * 16 + lan15;
      if (d_col < npos) {
        int dy = d_col / F;
        int dx = d_col - dy * F;
        int xo = flip_back ? (F - 1 - dx) : dx;
        float* op = outbase + (size_t)b * batch_stride + src_off +
                    (dy * F + xo) * Cout;
#pragma unroll
        for (int r = 0; r < 8; ++r) {
          int m = tm * 32 + mi * 16 + mrow + r;
          if (m < Cout) op[m] = acc[r] + bias[m];
        }
      }
    }
  }
}

// ---------------------------------------------------------------------------
// Weight pack: OIHW fp32 -> [Cout][ky][kx][Cin] f16
// ---------------------------------------------------------------------------
__global__ void pack_w_kernel(const float* __restrict__ w,
                              _Float16* __restrict__ wp,
                              int Cout, int Cin, int KHW)
{
  int i = blockIdx.x * blockDim.x + threadIdx.x;
  int total = Cout * Cin * KHW * KHW;
  if (i >= total) return;
  int ci = i % Cin;  int t = i / Cin;
  int kx = t % KHW;  t /= KHW;
  int ky = t % KHW;  int co = t / KHW;
  wp[i] = (_Float16)w[(((size_t)co * Cin + ci) * KHW + ky) * KHW + kx];
}

// ---------------------------------------------------------------------------
// Fused time-mean (T=4) + per-pixel L2-norm for feat1 -> NHWC f16.
// One 128-thread block per (b, pos); coalesced channel-contiguous writes.
// ---------------------------------------------------------------------------
__global__ __launch_bounds__(128) void mean1_l2norm_kernel(
    const float* __restrict__ f1, const float* __restrict__ l2w,
    _Float16* __restrict__ out)
{
  const int C = 480, HW = 38 * 38;
  const int pos = blockIdx.x;
  const int b   = blockIdx.y;

  float lm[4];
  int nc = 0;
  float local = 0.0f;
  for (int c = threadIdx.x; c < C; c += 128) {
    const float* p = f1 + ((size_t)(b * C + c) * 4) * HW + pos;
    float m = 0.25f * (p[0] + p[HW] + p[2 * HW] + p[3 * HW]);
    lm[nc++] = m;
    local += m * m;
  }
  __shared__ float red[128];
  red[threadIdx.x] = local;
  __syncthreads();
  for (int s = 64; s > 0; s >>= 1) {
    if (threadIdx.x < s) red[threadIdx.x] += red[threadIdx.x + s];
    __syncthreads();
  }
  const float inv = 1.0f / sqrtf(red[0] + 1e-10f);
  _Float16* op = out + ((size_t)b * HW + pos) * C;
  nc = 0;
  for (int c = threadIdx.x; c < C; c += 128) {
    op[c] = (_Float16)(lm[nc++] * inv * l2w[c]);
  }
}

// Time-mean (T=4) for feat2 -> NHWC f16 (i indexes output NHWC order).
__global__ void mean2_kernel(const float* __restrict__ f2,
                             _Float16* __restrict__ out, int total)
{
  int i = blockIdx.x * blockDim.x + threadIdx.x;
  if (i >= total) return;
  const int C = 832, HW = 19 * 19;
  int c = i % C;  int t = i / C;
  int pos = t % HW;  int b = t / HW;
  const float* p = f2 + ((size_t)(b * C + c) * 4) * HW + pos;
  out[i] = (_Float16)(0.25f * (p[0] + p[HW] + p[2 * HW] + p[3 * HW]));
}

// Row-wise softmax over 25 classes (safe for in == out).
__global__ void softmax25_kernel(const float* __restrict__ in,
                                 float* __restrict__ out, int rows)
{
  int r = blockIdx.x * blockDim.x + threadIdx.x;
  if (r >= rows) return;
  const float* p = in + (size_t)r * 25;
  float mx = p[0];
#pragma unroll
  for (int i = 1; i < 25; ++i) mx = fmaxf(mx, p[i]);
  float e[25], s = 0.0f;
#pragma unroll
  for (int i = 0; i < 25; ++i) { e[i] = __expf(p[i] - mx); s += e[i]; }
  const float inv = 1.0f / s;
  float* q = out + (size_t)r * 25;
#pragma unroll
  for (int i = 0; i < 25; ++i) q[i] = e[i] * inv;
}

// ---------------------------------------------------------------------------
// Host-side launchers
// ---------------------------------------------------------------------------
static inline void launch_conv(hipStream_t s, const _Float16* in, const _Float16* wp,
                               const float* bias, const _Float16* guard,
                               _Float16* out, int Cin, int Cout, int H, int W,
                               int Hout, int Wout, int k, int stride, int pad,
                               int flip_in)
{
  int npos  = Hout * Wout;
  int tiles = ((Cout + 31) / 32) * ((npos + 31) / 32);
  dim3 grid((tiles + 3) / 4, 16);
  dim3 block(128);
  if (k == 1)
    conv_relu_f16<1><<<grid, block, 0, s>>>(in, wp, bias, guard, out, Cin, Cout,
                                            H, W, Hout, Wout, stride, pad, flip_in);
  else
    conv_relu_f16<3><<<grid, block, 0, s>>>(in, wp, bias, guard, out, Cin, Cout,
                                            H, W, Hout, Wout, stride, pad, flip_in);
}

static inline void launch_head(hipStream_t s, const _Float16* in, const _Float16* wp,
                               const float* bias, const _Float16* guard,
                               float* outbase, int Cin, int Cout, int F,
                               int batch_stride, int src_off,
                               int flip_in, int flip_back)
{
  int npos  = F * F;
  int tiles = ((Cout + 31) / 32) * ((npos + 31) / 32);
  dim3 grid((tiles + 3) / 4, 16);
  dim3 block(128);
  head_conv_f32<<<grid, block, 0, s>>>(in, wp, bias, guard, outbase, Cin, Cout, F,
                                       batch_stride, src_off, flip_in, flip_back);
}

static inline void launch_pack(hipStream_t s, const float* w, _Float16* wp,
                               int Cout, int Cin, int KHW)
{
  int total = Cout * Cin * KHW * KHW;
  pack_w_kernel<<<(total + 255) / 256, 256, 0, s>>>(w, wp, Cout, Cin, KHW);
}

// ---------------------------------------------------------------------------
// kernel_launch
// Input order (setup_inputs dict order, lists flattened):
//   0: feat1  1: feat2  2: priors  3: l2norm_w
//   4..11: extras_w   12..19: extras_b
//   20..25: loc_w     26..31: loc_b   32..37: conf_w   38..43: conf_b
// Output: loc | conf_raw | priors | conf_sm | conf_flip | loc_flip (fp32)
// ---------------------------------------------------------------------------
extern "C" void kernel_launch(void* const* d_in, const int* in_sizes, int n_in,
                              void* d_out, int out_size, void* d_ws, size_t ws_size,
                              hipStream_t stream)
{
  (void)in_sizes; (void)n_in; (void)out_size; (void)ws_size;

  const float* feat1  = (const float*)d_in[0];
  const float* feat2  = (const float*)d_in[1];
  const float* priors = (const float*)d_in[2];
  const float* l2w    = (const float*)d_in[3];
  const float *exw[8], *exb[8];
  for (int i = 0; i < 8; ++i) { exw[i] = (const float*)d_in[4 + i];
                                exb[i] = (const float*)d_in[12 + i]; }
  const float *locw[6], *locb[6], *confw[6], *confb[6];
  for (int i = 0; i < 6; ++i) {
    locw[i]  = (const float*)d_in[20 + i];
    locb[i]  = (const float*)d_in[26 + i];
    confw[i] = (const float*)d_in[32 + i];
    confb[i] = (const float*)d_in[38 + i];
  }

  // ----- output slots -----
  const size_t NPRI = 8732;
  float* out    = (float*)d_out;
  float* o_loc  = out;                              // [16, 8732, 4]
  float* o_conf = o_loc  + (size_t)16 * NPRI * 4;   // [16, 8732, 25]
  float* o_pri  = o_conf + (size_t)16 * NPRI * 25;  // [8732, 4]
  float* o_sm   = o_pri  + NPRI * 4;                // [16, 8732, 25]
  float* o_cfl  = o_sm   + (size_t)16 * NPRI * 25;  // [16, 8732, 25]
  float* o_lfl  = o_cfl  + (size_t)16 * NPRI * 25;  // [16, 8732, 4]

  // ----- conv configs -----
  struct ExCfg { int ci, co, k, s, p, hin, win, hout, wout; };
  static const ExCfg EX[8] = {
    {832, 256, 1, 1, 0, 19, 19, 19, 19},
    {256, 512, 3, 2, 1, 19, 19, 10, 10},
    {512, 128, 1, 1, 0, 10, 10, 10, 10},
    {128, 256, 3, 2, 1, 10, 10,  5,  5},
    {256, 128, 1, 1, 0,  5,  5,  5,  5},
    {128, 256, 3, 1, 0,  5,  5,  3,  3},
    {256, 128, 1, 1, 0,  3,  3,  3,  3},
    {128, 256, 3, 1, 0,  3,  3,  1,  1},
  };
  static const int F[6]   = {38, 19, 10, 5, 3, 1};
  static const int CIN[6] = {480, 832, 512, 256, 256, 256};
  static const int MB[6]  = {4, 6, 6, 6, 4, 4};

  // ----- workspace layout (f16 halves) -----
  _Float16* ws = (_Float16*)d_ws;
  _Float16* guard = ws; ws += 1024;                     // zero pad row (2 KB)
  _Float16* m1 = ws; ws += (size_t)16 * 38 * 38 * 480;  // NHWC l2norm'd mean
  _Float16* m2 = ws; ws += (size_t)16 * 19 * 19 * 832;  // NHWC mean
  static const int esz[8] = {361 * 256, 100 * 512, 100 * 128, 25 * 256,
                             25 * 128,  9 * 256,   9 * 128,   1 * 256};
  _Float16* e[2][8];
  for (int p = 0; p < 2; ++p)
    for (int i = 0; i < 8; ++i) { e[p][i] = ws; ws += (size_t)16 * esz[i]; }
  _Float16* wp_ex[8];
  for (int i = 0; i < 8; ++i) {
    wp_ex[i] = ws; ws += (size_t)EX[i].co * EX[i].ci * EX[i].k * EX[i].k;
  }
  _Float16 *wp_loc[6], *wp_conf[6];
  for (int i = 0; i < 6; ++i) {
    wp_loc[i]  = ws; ws += (size_t)(MB[i] * 4)  * CIN[i] * 9;
    wp_conf[i] = ws; ws += (size_t)(MB[i] * 25) * CIN[i] * 9;
  }

  // ----- guard zero-fill + weight packing + time-means -----
  hipMemsetAsync(guard, 0, 1024 * sizeof(_Float16), stream);
  for (int i = 0; i < 8; ++i)
    launch_pack(stream, exw[i], wp_ex[i], EX[i].co, EX[i].ci, EX[i].k);
  for (int i = 0; i < 6; ++i) {
    launch_pack(stream, locw[i],  wp_loc[i],  MB[i] * 4,  CIN[i], 3);
    launch_pack(stream, confw[i], wp_conf[i], MB[i] * 25, CIN[i], 3);
  }
  mean1_l2norm_kernel<<<dim3(38 * 38, 16), 128, 0, stream>>>(feat1, l2w, m1);
  {
    int total = 16 * 19 * 19 * 832;
    mean2_kernel<<<(total + 255) / 256, 256, 0, stream>>>(feat2, m2, total);
  }

  // ----- extras chain, both paths (path 1 flips the shared input once) -----
  for (int p = 0; p < 2; ++p) {
    const _Float16* cur = m2;
    for (int i = 0; i < 8; ++i) {
      int flip = (p == 1 && i == 0) ? 1 : 0;
      launch_conv(stream, cur, wp_ex[i], exb[i], guard, e[p][i],
                  EX[i].ci, EX[i].co, EX[i].hin, EX[i].win,
                  EX[i].hout, EX[i].wout, EX[i].k, EX[i].s, EX[i].p, flip);
      cur = e[p][i];
    }
  }

  // ----- heads -----
  int loc_off[6], conf_off[6];
  {
    int lo = 0, co = 0;
    for (int i = 0; i < 6; ++i) {
      loc_off[i] = lo;  conf_off[i] = co;
      lo += F[i] * F[i] * MB[i] * 4;
      co += F[i] * F[i] * MB[i] * 25;
    }
  }
  for (int p = 0; p < 2; ++p) {
    const _Float16* srcs[6] = {m1, m2, e[p][1], e[p][3], e[p][5], e[p][7]};
    float* lbase = p ? o_lfl : o_loc;
    float* cbase = p ? o_cfl : o_conf;
    for (int i = 0; i < 6; ++i) {
      int flip_in   = (p == 1 && i < 2) ? 1 : 0;  // shared sources flip-gather
      int flip_back = p;
      launch_head(stream, srcs[i], wp_loc[i], locb[i], guard, lbase,
                  CIN[i], MB[i] * 4,  F[i], (int)(NPRI * 4),  loc_off[i],
                  flip_in, flip_back);
      launch_head(stream, srcs[i], wp_conf[i], confb[i], guard, cbase,
                  CIN[i], MB[i] * 25, F[i], (int)(NPRI * 25), conf_off[i],
                  flip_in, flip_back);
    }
  }

  // ----- priors pass-through (capture-safe D2D copy) -----
  hipMemcpyAsync(o_pri, priors, NPRI * 4 * sizeof(float),
                 hipMemcpyDeviceToDevice, stream);

  // ----- softmaxes (conf_sm from conf_raw; conf_flip in place) -----
  {
    int rows = 16 * (int)NPRI;
    softmax25_kernel<<<(rows + 255) / 256, 256, 0, stream>>>(o_conf, o_sm, rows);
    softmax25_kernel<<<(rows + 255) / 256, 256, 0, stream>>>(o_cfl, o_cfl, rows);
  }
}
